// DockingAwareAttention_85925115723855
// MI455X (gfx1250) — compile-verified, matrix operand order checked by reading, not run on hardware
//
#include <hip/hip_runtime.h>
#include <hip/hip_bf16.h>

typedef _Float16 f16;
typedef _Float16 v16h __attribute__((ext_vector_type(16)));
typedef _Float16 v8h  __attribute__((ext_vector_type(8)));
typedef float    v8f  __attribute__((ext_vector_type(8)));

#define D_MODEL   1024
#define NUM_HEADS 16
#define HEAD_DIM  64
#define SEQ       2048
#define BATCH     2
#define MTOT      (BATCH*SEQ)      // 4096 rows in every GEMM

union Frag16 { v16h v; v8h h[2]; };

__device__ __forceinline__ v8f wmma16(v16h a, v16h b, v8f c) {
  // emits v_wmma_f32_16x16x32_f16
  return __builtin_amdgcn_wmma_f32_16x16x32_f16(false, a, false, b, (short)0, c, false, false);
}

__device__ __forceinline__ Frag16 load_frag(const f16* p) {
  Frag16 f;
  f.h[0] = *(const v8h*)(p);       // K 0..7  (lanes<16) / 8..15  (lanes>=16)
  f.h[1] = *(const v8h*)(p + 16);  // K 16..23 / 24..31
  return f;
}

// DPP16 row_xmask: lane <-> lane^MASK within each row of 16 lanes (pure VALU,
// no LDS pipeline, no dscnt waits -- replaces ds_bpermute shuffles).
template<int MASK>
__device__ __forceinline__ float xor16(float v) {
  int r = __builtin_amdgcn_update_dpp(0, __builtin_bit_cast(int, v),
                                      0x160 | MASK, 0xf, 0xf, true);
  return __builtin_bit_cast(float, r);
}
__device__ __forceinline__ float rowmax16(float v) {
  v = fmaxf(v, xor16<1>(v));
  v = fmaxf(v, xor16<2>(v));
  v = fmaxf(v, xor16<4>(v));
  v = fmaxf(v, xor16<8>(v));
  return v;
}
__device__ __forceinline__ float rowsum16(float v) {
  v += xor16<1>(v);
  v += xor16<2>(v);
  v += xor16<4>(v);
  v += xor16<8>(v);
  return v;
}

// ---------------------------------------------------------------------------
// fp32 -> fp16 conversion of activations
__global__ __launch_bounds__(256)
void cvt_x_kernel(const float* __restrict__ x, f16* __restrict__ xh, int n) {
  int i = blockIdx.x * 256 + threadIdx.x;
  if (i < n) xh[i] = (f16)x[i];
}

// fp32 W[k][n] -> fp16 Wt[n][k]  (B-fragment friendly layout)
__global__ __launch_bounds__(256)
void cvt_wT_kernel(const float* __restrict__ w, f16* __restrict__ wt) {
  int i = blockIdx.x * 256 + threadIdx.x;   // i = n*1024 + k
  int n = i >> 10, k = i & 1023;
  wt[i] = (f16)w[(size_t)k * D_MODEL + n];
}

// ---------------------------------------------------------------------------
// WMMA GEMM: Y[4096,1024] = A[4096,1024] @ W (W transposed as Bt[n][k]) + bias
// Register-double-buffered pipeline: next tile's global loads are in flight
// while the current tile's fragments/WMMAs execute.
// MODE 0: f32 row-major output (final O-projection)
// MODE 1: f16 output into [B,H,S,64] with (acc+bias)*scale   (Q pre-scaled, K)
// MODE 2: f16 output into [B,H,64,S] (V transposed for PV WMMA)
template<int MODE>
__global__ __launch_bounds__(256)
void gemm_wmma_kernel(const f16* __restrict__ A,
                      const f16* __restrict__ Bt,
                      const float* __restrict__ bias,
                      void* __restrict__ dstv,
                      float scale)
{
  __shared__ f16 a_lds[128 * 40];   // 128 rows x 32 halves, pad->40
  __shared__ f16 b_lds[128 * 40];

  const int t    = threadIdx.x;
  const int lane = t & 31;
  const int wid  = t >> 5;
  const int lr   = lane & 15;
  const int hl   = lane >> 4;

  const int n0 = blockIdx.x * 128;
  const int m0 = blockIdx.y * 128;
  const int wm = (wid & 1) * 64;    // 2 waves along M -> 64 rows each
  const int wn = (wid >> 1) * 32;   // 4 waves along N -> 32 cols each

  // staging geometry: 512 16B slots per tile, 2 per thread
  int row_[2], seg_[2];
#pragma unroll
  for (int i = 0; i < 2; ++i) { int s = t + i * 256; row_[i] = s >> 2; seg_[i] = s & 3; }

  const v8f zero8 = {0.f,0.f,0.f,0.f,0.f,0.f,0.f,0.f};
  v8f acc[4][2];
#pragma unroll
  for (int mt = 0; mt < 4; ++mt)
#pragma unroll
    for (int nt = 0; nt < 2; ++nt) acc[mt][nt] = zero8;

  uint4 ra[2], rb[2];
#pragma unroll
  for (int i = 0; i < 2; ++i) {   // prologue: tile k0=0 -> registers
    ra[i] = *(const uint4*)(A  + (size_t)(m0 + row_[i]) * D_MODEL + seg_[i] * 8);
    rb[i] = *(const uint4*)(Bt + (size_t)(n0 + row_[i]) * D_MODEL + seg_[i] * 8);
  }

  for (int k0 = 0; k0 < D_MODEL; k0 += 32) {
    __syncthreads();
#pragma unroll
    for (int i = 0; i < 2; ++i) {
      *(uint4*)&a_lds[row_[i] * 40 + seg_[i] * 8] = ra[i];
      *(uint4*)&b_lds[row_[i] * 40 + seg_[i] * 8] = rb[i];
    }
    __syncthreads();

    if (k0 + 32 < D_MODEL) {      // issue next tile's loads; latency overlaps WMMAs
#pragma unroll
      for (int i = 0; i < 2; ++i) {
        ra[i] = *(const uint4*)(A  + (size_t)(m0 + row_[i]) * D_MODEL + k0 + 32 + seg_[i] * 8);
        rb[i] = *(const uint4*)(Bt + (size_t)(n0 + row_[i]) * D_MODEL + k0 + 32 + seg_[i] * 8);
      }
    }

    Frag16 af[4], bf[2];
#pragma unroll
    for (int mt = 0; mt < 4; ++mt)
      af[mt] = load_frag(&a_lds[(wm + mt * 16 + lr) * 40 + hl * 8]);
#pragma unroll
    for (int nt = 0; nt < 2; ++nt)
      bf[nt] = load_frag(&b_lds[(wn + nt * 16 + lr) * 40 + hl * 8]);

#pragma unroll
    for (int mt = 0; mt < 4; ++mt)
#pragma unroll
      for (int nt = 0; nt < 2; ++nt)
        acc[mt][nt] = wmma16(af[mt].v, bf[nt].v, acc[mt][nt]);
  }

  // epilogue: C layout -> lane holds col n=lr, rows r + 8*hl
#pragma unroll
  for (int mt = 0; mt < 4; ++mt) {
#pragma unroll
    for (int nt = 0; nt < 2; ++nt) {
      int n = n0 + wn + nt * 16 + lr;
      float bsv = bias[n];
#pragma unroll
      for (int r = 0; r < 8; ++r) {
        int m = m0 + wm + mt * 16 + r + 8 * hl;
        float val = acc[mt][nt][r] + bsv;
        if (MODE == 0) {
          ((float*)dstv)[(size_t)m * D_MODEL + n] = val;
        } else {
          int b = m >> 11, s = m & (SEQ - 1);
          int h = n >> 6, dh = n & 63;
          f16* d16 = (f16*)dstv;
          if (MODE == 1)
            d16[(((size_t)(b * NUM_HEADS + h)) * SEQ + s) * HEAD_DIM + dh] =
              (f16)(val * scale);
          else  // MODE 2: V transposed [B,H,64,S]
            d16[(((size_t)(b * NUM_HEADS + h)) * HEAD_DIM + dh) * SEQ + s] =
              (f16)val;
        }
      }
    }
  }
}

// ---------------------------------------------------------------------------
// dsV[b,h,d] = sum_s ds[b,s] * V[b,h,s,d]   (rank-1 docking correction)
__global__ __launch_bounds__(256)
void dsv_kernel(const float* __restrict__ ds, const f16* __restrict__ Vt,
                float* __restrict__ dsv)
{
  int idx = blockIdx.x * 256 + threadIdx.x;   // [B,H,64] = 2048
  if (idx >= BATCH * NUM_HEADS * HEAD_DIM) return;
  int b = idx / (NUM_HEADS * HEAD_DIM);
  const f16*   v = Vt + (size_t)idx * SEQ;    // Vt row is contiguous over s
  const float* d = ds + (size_t)b * SEQ;
  float acc = 0.f;
  for (int s = 0; s < SEQ; ++s) acc += d[s] * (float)v[s];
  dsv[idx] = acc;
}

// ---------------------------------------------------------------------------
// Flash attention + docking correction.
// grid (S/128, H, B), 256 threads (8 waves); each wave owns 16 q rows.
// K/V chunks are register-double-buffered; softmax stats use DPP row_xmask.
__global__ __launch_bounds__(256)
void attn_kernel(const f16* __restrict__ Q,    // [B,H,S,64] pre-scaled 1/8
                 const f16* __restrict__ K,    // [B,H,S,64]
                 const f16* __restrict__ Vt,   // [B,H,64,S]
                 const float* __restrict__ dsv,// [B,H,64]
                 const float* __restrict__ alphap,
                 f16* __restrict__ ctx)        // [B,S,H*64]
{
  __shared__ f16 kbuf[32 * 72];        // 32 keys x 64 d (+pad)
  __shared__ f16 vbuf[64 * 40];        // 64 d x 32 keys (+pad)
  __shared__ f16 pbuf[8 * 16 * 40];    // per-wave P re-layout buffer

  const int t = threadIdx.x, lane = t & 31, wid = t >> 5;
  const int lr = lane & 15, hl = lane >> 4;
  const int h = blockIdx.y, b = blockIdx.z;
  const int q0 = blockIdx.x * 128 + wid * 16;
  const float alpha = *alphap;

  const f16* Qb = Q  + (((size_t)(b * NUM_HEADS + h)) * SEQ + q0) * HEAD_DIM;
  const f16* Kb = K  +  ((size_t)(b * NUM_HEADS + h)) * SEQ * HEAD_DIM;
  const f16* Vb = Vt +  ((size_t)(b * NUM_HEADS + h)) * HEAD_DIM * SEQ;
  f16* pw = &pbuf[wid * 16 * 40];

  // staging geometry (one 16B slot per thread per buffer)
  const int kkey = t >> 3, ksg = t & 7;     // K: 32 keys x 8 segs
  const int vd   = t >> 2, vsg = t & 3;     // V: 64 d    x 4 segs

  // Q A-fragments, d-chunks [0,32) and [32,64)
  Frag16 aq0, aq1;
  aq0.h[0] = *(const v8h*)(Qb + lr * 64 + hl * 8);
  aq0.h[1] = *(const v8h*)(Qb + lr * 64 + hl * 8 + 16);
  aq1.h[0] = *(const v8h*)(Qb + lr * 64 + 32 + hl * 8);
  aq1.h[1] = *(const v8h*)(Qb + lr * 64 + 32 + hl * 8 + 16);

  const v8f zero8 = {0.f,0.f,0.f,0.f,0.f,0.f,0.f,0.f};
  float mrow[8], lsum[8];
  v8f oacc[4];
#pragma unroll
  for (int r = 0; r < 8; ++r) { mrow[r] = -3.0e38f; lsum[r] = 0.f; }
#pragma unroll
  for (int dt = 0; dt < 4; ++dt) oacc[dt] = zero8;

  // prologue: chunk 0 -> registers
  uint4 rk = *(const uint4*)(Kb + (size_t)kkey * 64 + ksg * 8);
  uint4 rv = *(const uint4*)(Vb + (size_t)vd * SEQ + vsg * 8);

  for (int c = 0; c < SEQ / 32; ++c) {
    __syncthreads();
    *(uint4*)&kbuf[kkey * 72 + ksg * 8] = rk;
    *(uint4*)&vbuf[vd * 40 + vsg * 8]   = rv;
    __syncthreads();

    if (c + 1 < SEQ / 32) {   // next chunk's loads overlap this chunk's compute
      const int kg = (c + 1) * 32;
      rk = *(const uint4*)(Kb + (size_t)(kg + kkey) * 64 + ksg * 8);
      rv = *(const uint4*)(Vb + (size_t)vd * SEQ + kg + vsg * 8);
    }

    // scores: two 16-key tiles, K-dim = 64 -> 2 WMMA each
    v8f cs[2];
#pragma unroll
    for (int t2 = 0; t2 < 2; ++t2) {
      Frag16 bk0 = load_frag(&kbuf[(t2 * 16 + lr) * 72 + hl * 8]);
      Frag16 bk1 = load_frag(&kbuf[(t2 * 16 + lr) * 72 + 32 + hl * 8]);
      cs[t2] = wmma16(aq0.v, bk0.v, zero8);
      cs[t2] = wmma16(aq1.v, bk1.v, cs[t2]);
    }

    // online softmax stats + P store to LDS (C layout: col=lr, row=r+8*hl)
#pragma unroll
    for (int r = 0; r < 8; ++r) {
      float vmax = rowmax16(fmaxf(cs[0][r], cs[1][r]));
      float mnew = fmaxf(mrow[r], vmax);
      float fcor = __expf(mrow[r] - mnew);
      float p0 = __expf(cs[0][r] - mnew);
      float p1 = __expf(cs[1][r] - mnew);
      lsum[r] = lsum[r] * fcor + rowsum16(p0 + p1);
      mrow[r] = mnew;
#pragma unroll
      for (int dt = 0; dt < 4; ++dt) oacc[dt][r] *= fcor;
      pw[(r + 8 * hl) * 40 + lr]      = (f16)p0;
      pw[(r + 8 * hl) * 40 + 16 + lr] = (f16)p1;
    }

    // P @ V : A = P (16x32 keys), B = Vt rows (d in lanes, keys striped)
    Frag16 ap = load_frag(&pw[lr * 40 + hl * 8]);
#pragma unroll
    for (int dt = 0; dt < 4; ++dt) {
      Frag16 bv = load_frag(&vbuf[(dt * 16 + lr) * 40 + hl * 8]);
      oacc[dt] = wmma16(ap.v, bv.v, oacc[dt]);
    }
  }

  // epilogue: ctx = (1-a)*softmax_ctx + a*dsV, stored f16 [B,S,H*64]
  float inv[8];
#pragma unroll
  for (int r = 0; r < 8; ++r) inv[r] = (1.0f - alpha) / lsum[r];
#pragma unroll
  for (int dt = 0; dt < 4; ++dt) {
    float dv = alpha * dsv[(b * NUM_HEADS + h) * HEAD_DIM + dt * 16 + lr];
#pragma unroll
    for (int r = 0; r < 8; ++r) {
      int s = q0 + r + 8 * hl;
      ctx[((size_t)(b * SEQ + s)) * D_MODEL + h * 64 + dt * 16 + lr] =
        (f16)(oacc[dt][r] * inv[r] + dv);
    }
  }
}

// ---------------------------------------------------------------------------
extern "C" void kernel_launch(void* const* d_in, const int* in_sizes, int n_in,
                              void* d_out, int out_size, void* d_ws, size_t ws_size,
                              hipStream_t stream) {
  const float* x     = (const float*)d_in[0];
  const float* ds    = (const float*)d_in[1];
  const float* q_w   = (const float*)d_in[2];
  const float* q_b   = (const float*)d_in[3];
  const float* k_w   = (const float*)d_in[4];
  const float* k_b   = (const float*)d_in[5];
  const float* v_w   = (const float*)d_in[6];
  const float* v_b   = (const float*)d_in[7];
  const float* o_w   = (const float*)d_in[8];
  const float* o_b   = (const float*)d_in[9];
  const float* alpha = (const float*)d_in[10];
  float* out = (float*)d_out;

  char* ws = (char*)d_ws;
  size_t off = 0;
  auto walloc = [&](size_t bytes) -> void* {
    void* p = ws + off; off += (bytes + 255) & ~(size_t)255; return p;
  };
  f16* xh   = (f16*)walloc((size_t)MTOT * D_MODEL * 2);      // x in f16
  f16* qwt  = (f16*)walloc((size_t)D_MODEL * D_MODEL * 2);
  f16* kwt  = (f16*)walloc((size_t)D_MODEL * D_MODEL * 2);
  f16* vwt  = (f16*)walloc((size_t)D_MODEL * D_MODEL * 2);
  f16* owt  = (f16*)walloc((size_t)D_MODEL * D_MODEL * 2);
  f16* Qh   = (f16*)walloc((size_t)MTOT * D_MODEL * 2);      // [B,H,S,64]
  f16* Kh   = (f16*)walloc((size_t)MTOT * D_MODEL * 2);      // [B,H,S,64]
  f16* Vth  = (f16*)walloc((size_t)MTOT * D_MODEL * 2);      // [B,H,64,S]
  f16* ctxh = (f16*)walloc((size_t)MTOT * D_MODEL * 2);      // [B,S,1024]
  float* dsv = (float*)walloc((size_t)BATCH * NUM_HEADS * HEAD_DIM * 4);

  const int nElem = MTOT * D_MODEL;
  cvt_x_kernel<<<(nElem + 255) / 256, 256, 0, stream>>>(x, xh, nElem);
  const int nW = D_MODEL * D_MODEL;
  cvt_wT_kernel<<<nW / 256, 256, 0, stream>>>(q_w, qwt);
  cvt_wT_kernel<<<nW / 256, 256, 0, stream>>>(k_w, kwt);
  cvt_wT_kernel<<<nW / 256, 256, 0, stream>>>(v_w, vwt);
  cvt_wT_kernel<<<nW / 256, 256, 0, stream>>>(o_w, owt);

  dim3 ggrid(D_MODEL / 128, MTOT / 128);   // (8, 32)
  gemm_wmma_kernel<1><<<ggrid, 256, 0, stream>>>(xh, qwt, q_b, Qh, 0.125f);
  gemm_wmma_kernel<1><<<ggrid, 256, 0, stream>>>(xh, kwt, k_b, Kh, 1.0f);
  gemm_wmma_kernel<2><<<ggrid, 256, 0, stream>>>(xh, vwt, v_b, Vth, 1.0f);

  dsv_kernel<<<(BATCH * NUM_HEADS * HEAD_DIM + 255) / 256, 256, 0, stream>>>(ds, Vth, dsv);

  dim3 agrid(SEQ / 128, NUM_HEADS, BATCH); // (16, 16, 2)
  attn_kernel<<<agrid, 256, 0, stream>>>(Qh, Kh, Vth, dsv, alpha, ctxh);

  gemm_wmma_kernel<0><<<ggrid, 256, 0, stream>>>(ctxh, owt, o_b, out, 1.0f);
}
